// LSA_2147483648500
// MI455X (gfx1250) — matrix operand order repaired and checked by reading, not compiled, not run on hardware
//
#include <hip/hip_runtime.h>
#include <hip/hip_bf16.h>

// ---------------------------------------------------------------------------
// x[4,2048,1024] fp32 -> attention (16 heads x 64) with masked diagonal ->
// out-projection. All GEMMs in bf16 WMMA (v_wmma_f32_16x16x32_bf16), fp32 acc.
// ---------------------------------------------------------------------------
#define BATCH   4
#define SEQ     2048
#define DMODEL  1024
#define HEADS   16
#define DHEAD   64
#define INNER   1024      // HEADS*DHEAD
#define E3      3072      // 3*INNER
#define ROWS    (BATCH*SEQ)   // 8192

typedef __attribute__((ext_vector_type(16))) __bf16 v16bf;
typedef __attribute__((ext_vector_type(8)))  __bf16 bf16x8;
typedef __attribute__((ext_vector_type(2)))  __bf16 bf16x2;
typedef __attribute__((ext_vector_type(8)))  float  v8f;

__device__ __forceinline__ v8f wmma_bf16(v16bf a, v16bf b, v8f c) {
  return __builtin_amdgcn_wmma_f32_16x16x32_bf16(
      false, a, false, b, (short)0, c, false, false);
}

// A fragment (16x32 bf16, row-major source, leading dim ld):
// lanes 0-15: M=lane,    elems 0-7 -> K=k0+0..7,  elems 8-15 -> K=k0+16..23
// lanes16-31: M=lane-16, elems 0-7 -> K=k0+8..15, elems 8-15 -> K=k0+24..31
__device__ __forceinline__ v16bf load_fragA(const __bf16* X, int ld, int m0, int k0) {
  int lane = threadIdx.x & 31;
  int hi   = lane >> 4;
  const __bf16* p = X + (size_t)(m0 + (lane & 15)) * ld + k0 + hi * 8;
  v16bf a;
#pragma unroll
  for (int v = 0; v < 4; ++v) {
    bf16x2 t = *(const bf16x2*)(p + 2 * v);
    a[2 * v] = t[0]; a[2 * v + 1] = t[1];
  }
#pragma unroll
  for (int v = 0; v < 4; ++v) {
    bf16x2 t = *(const bf16x2*)(p + 16 + 2 * v);
    a[8 + 2 * v] = t[0]; a[8 + 2 * v + 1] = t[1];
  }
  return a;
}

// B fragment (32x16 bf16) from transposed (N-major) BT[n][k], leading dim ld:
// lanes 0-15: N=n0+lane, K=k0+0..7,16..23 ; lanes 16-31: N=n0+lane-16, K=k0+8..15,24..31
__device__ __forceinline__ v16bf load_fragB_T(const __bf16* BT, int ld, int k0, int n0) {
  int lane = threadIdx.x & 31;
  int hi   = lane >> 4;
  const __bf16* p = BT + (size_t)(n0 + (lane & 15)) * ld + k0 + hi * 8;
  v16bf b;
#pragma unroll
  for (int v = 0; v < 8; ++v) {
    bf16x2 t = *(const bf16x2*)(p + 2 * v + (v >> 2) * 8);
    b[2 * v] = t[0]; b[2 * v + 1] = t[1];
  }
  return b;
}

// ---------------------------------------------------------------------------
// 1) fp32 -> bf16 converters
// ---------------------------------------------------------------------------
__global__ void __launch_bounds__(256) cvt_bf16_kernel(const float* __restrict__ src,
                                                       __bf16* __restrict__ dst, int n) {
  for (int i = blockIdx.x * blockDim.x + threadIdx.x; i < n; i += gridDim.x * blockDim.x)
    dst[i] = (__bf16)src[i];
}

// src[rows][cols] (row-major) -> dst[cols][rows]
__global__ void __launch_bounds__(256) cvt_bf16_T_kernel(const float* __restrict__ src,
                                                         __bf16* __restrict__ dst,
                                                         int rows, int cols) {
  int n = rows * cols;
  for (int i = blockIdx.x * blockDim.x + threadIdx.x; i < n; i += gridDim.x * blockDim.x) {
    int r = i / cols, c = i - r * cols;
    dst[(size_t)c * rows + r] = (__bf16)src[i];
  }
}

// ---------------------------------------------------------------------------
// 2) QKV projection: 32x64 output tile per wave (2 A-frags x 4 B-frags,
//    8 WMMA per k-step). Q scaled by exp(temperature), stored [b,h,n,64];
//    K stored [b,h,n,64]; V stored transposed [b,h,64,n].
// ---------------------------------------------------------------------------
__global__ void __launch_bounds__(256) qkv_gemm_kernel(
    const __bf16* __restrict__ xb, const __bf16* __restrict__ wqkvT,
    const float* __restrict__ temperature,
    __bf16* __restrict__ qb, __bf16* __restrict__ kb, __bf16* __restrict__ vT) {
  const int wid    = blockIdx.x * (blockDim.x >> 5) + (threadIdx.x >> 5);
  const int tile_n = wid % (E3 / 64);
  const int tile_m = wid / (E3 / 64);
  const int m0 = tile_m * 32, n0 = tile_n * 64;
  const int lane = threadIdx.x & 31, hi = lane >> 4, nn = lane & 15;

  v8f acc[2][4];
#pragma unroll
  for (int mi = 0; mi < 2; ++mi)
#pragma unroll
    for (int f = 0; f < 4; ++f)
      acc[mi][f] = (v8f){0.f,0.f,0.f,0.f,0.f,0.f,0.f,0.f};

#pragma unroll 2
  for (int kk = 0; kk < DMODEL; kk += 32) {
    v16bf a0 = load_fragA(xb, DMODEL, m0, kk);
    v16bf a1 = load_fragA(xb, DMODEL, m0 + 16, kk);
    v16bf b0 = load_fragB_T(wqkvT, DMODEL, kk, n0);
    v16bf b1 = load_fragB_T(wqkvT, DMODEL, kk, n0 + 16);
    v16bf b2 = load_fragB_T(wqkvT, DMODEL, kk, n0 + 32);
    v16bf b3 = load_fragB_T(wqkvT, DMODEL, kk, n0 + 48);
    acc[0][0] = wmma_bf16(a0, b0, acc[0][0]);
    acc[0][1] = wmma_bf16(a0, b1, acc[0][1]);
    acc[0][2] = wmma_bf16(a0, b2, acc[0][2]);
    acc[0][3] = wmma_bf16(a0, b3, acc[0][3]);
    acc[1][0] = wmma_bf16(a1, b0, acc[1][0]);
    acc[1][1] = wmma_bf16(a1, b1, acc[1][1]);
    acc[1][2] = wmma_bf16(a1, b2, acc[1][2]);
    acc[1][3] = wmma_bf16(a1, b3, acc[1][3]);
  }

  const float scale = __expf(temperature[0]);
  const int which  = n0 >> 10;              // 64-col span never straddles q/k/v
  const int inner0 = n0 & (INNER - 1);
  const int h      = inner0 >> 6;           // 64-col span = exactly one head
  const float sc   = (which == 0) ? scale : 1.0f;

#pragma unroll
  for (int mi = 0; mi < 2; ++mi) {
    const int row0 = m0 + mi * 16;
    const int bidx = row0 >> 11;
    const int nrow0 = row0 & (SEQ - 1);
    const size_t bh = (size_t)(bidx * HEADS + h);
#pragma unroll
    for (int f = 0; f < 4; ++f) {
      const int dh0 = f * 16;
#pragma unroll
      for (int r = 0; r < 8; ++r) {
        int nrow = nrow0 + r + hi * 8;
        int colh = dh0 + nn;
        __bf16 val = (__bf16)(acc[mi][f][r] * sc);
        if (which == 0)      qb[(bh * SEQ + nrow) * DHEAD + colh] = val;
        else if (which == 1) kb[(bh * SEQ + nrow) * DHEAD + colh] = val;
        else                 vT[(bh * DHEAD + colh) * SEQ + nrow] = val;
      }
    }
  }
}

// ---------------------------------------------------------------------------
// 3) Attention: per wave one (b,h, 16-row query tile); stream 64 blocks of
//    32 keys. 4 WMMA for S (two 16-col frags), masked diag, online softmax,
//    P(16x32) staged in LDS -> full-K A frag, 4 full-K WMMA for P@V.
// ---------------------------------------------------------------------------
__global__ void __launch_bounds__(256) attn_kernel(
    const __bf16* __restrict__ qb, const __bf16* __restrict__ kb,
    const __bf16* __restrict__ vT, __bf16* __restrict__ ctx) {
  __shared__ __align__(16) __bf16 pbuf[8][512];   // one 16x32 P tile per wave

  const int wv  = threadIdx.x >> 5;
  const int wid = blockIdx.x * (blockDim.x >> 5) + wv;
  const int bh  = wid >> 7;
  const int q0  = (wid & 127) * 16;
  const int lane = threadIdx.x & 31, hi = lane >> 4, nn = lane & 15;

  const __bf16* Q = qb + (size_t)bh * SEQ * DHEAD;
  const __bf16* K = kb + (size_t)bh * SEQ * DHEAD;
  const __bf16* V = vT + (size_t)bh * DHEAD * SEQ;
  __bf16* pls = &pbuf[wv][0];

  v16bf a0 = load_fragA(Q, DHEAD, q0, 0);
  v16bf a1 = load_fragA(Q, DHEAD, q0, 32);

  float mrun[8], lrun[8];
  v8f o0 = {0.f,0.f,0.f,0.f,0.f,0.f,0.f,0.f}, o1 = o0, o2 = o0, o3 = o0;
#pragma unroll
  for (int r = 0; r < 8; ++r) { mrun[r] = -1e30f; lrun[r] = 0.f; }

  for (int jt = 0; jt < SEQ / 32; ++jt) {
    const int key0 = jt * 32;
    // ---- S = Q @ K^T for 32 keys: two 16x16 score frags
    v8f sl = {0.f,0.f,0.f,0.f,0.f,0.f,0.f,0.f}, sh = sl;
    sl = wmma_bf16(a0, load_fragB_T(K, DHEAD, 0,  key0), sl);
    sl = wmma_bf16(a1, load_fragB_T(K, DHEAD, 32, key0), sl);
    sh = wmma_bf16(a0, load_fragB_T(K, DHEAD, 0,  key0 + 16), sh);
    sh = wmma_bf16(a1, load_fragB_T(K, DHEAD, 32, key0 + 16), sh);

    // ---- online softmax over 32 columns (rows in 16-lane halves, M=r+8*hi)
#pragma unroll
    for (int r = 0; r < 8; ++r) {
      float svl = sl[r], svh = sh[r];
      const int qrow = q0 + r + hi * 8;
      if (qrow == key0 + nn)      svl = -1e30f;   // masked diagonal
      if (qrow == key0 + 16 + nn) svh = -1e30f;
      float mx = fmaxf(svl, svh);
      mx = fmaxf(mx, __shfl_xor(mx, 1));
      mx = fmaxf(mx, __shfl_xor(mx, 2));
      mx = fmaxf(mx, __shfl_xor(mx, 4));
      mx = fmaxf(mx, __shfl_xor(mx, 8));
      float mnew = fmaxf(mrun[r], mx);
      float corr = __expf(mrun[r] - mnew);
      float pl = __expf(svl - mnew);
      float ph = __expf(svh - mnew);
      float rs = pl + ph;
      rs += __shfl_xor(rs, 1);
      rs += __shfl_xor(rs, 2);
      rs += __shfl_xor(rs, 4);
      rs += __shfl_xor(rs, 8);
      lrun[r] = lrun[r] * corr + rs;
      mrun[r] = mnew;
      o0[r] *= corr; o1[r] *= corr; o2[r] *= corr; o3[r] *= corr;
      pls[(r + hi * 8) * 32 + nn]      = (__bf16)pl;
      pls[(r + hi * 8) * 32 + 16 + nn] = (__bf16)ph;
    }

    asm volatile("s_wait_dscnt 0" ::: "memory");   // LDS write->read, same wave

    // ---- re-fragment P (16x32) as WMMA A: two aligned 16B LDS reads
    bf16x8 t0 = *(const bf16x8*)(pls + (lane & 15) * 32 + hi * 8);
    bf16x8 t1 = *(const bf16x8*)(pls + (lane & 15) * 32 + 16 + hi * 8);
    v16bf pa;
#pragma unroll
    for (int i = 0; i < 8; ++i) { pa[i] = t0[i]; pa[8 + i] = t1[i]; }

    // ---- O += P @ V  (B(k=seq, n=dh) = vT[dh][key0+k], full 32 lanes real)
    o0 = wmma_bf16(pa, load_fragB_T(V, SEQ, key0, 0),  o0);
    o1 = wmma_bf16(pa, load_fragB_T(V, SEQ, key0, 16), o1);
    o2 = wmma_bf16(pa, load_fragB_T(V, SEQ, key0, 32), o2);
    o3 = wmma_bf16(pa, load_fragB_T(V, SEQ, key0, 48), o3);
  }

  // ---- epilogue: O / l, write context [b, n, h*64 + dh] in bf16
  const int bidx = bh >> 4, h = bh & 15;
#pragma unroll
  for (int r = 0; r < 8; ++r) {
    int M = r + hi * 8;
    float inv = 1.0f / lrun[r];
    size_t base = ((size_t)bidx * SEQ + q0 + M) * INNER + h * DHEAD + nn;
    ctx[base +  0] = (__bf16)(o0[r] * inv);
    ctx[base + 16] = (__bf16)(o1[r] * inv);
    ctx[base + 32] = (__bf16)(o2[r] * inv);
    ctx[base + 48] = (__bf16)(o3[r] * inv);
  }
}

// ---------------------------------------------------------------------------
// 4) Output projection: 32x64 tile per wave; out = ctx @ W_out + b_out (fp32)
// ---------------------------------------------------------------------------
__global__ void __launch_bounds__(256) out_gemm_kernel(
    const __bf16* __restrict__ ctx, const __bf16* __restrict__ woutT,
    const float* __restrict__ b_out, float* __restrict__ out) {
  const int wid    = blockIdx.x * (blockDim.x >> 5) + (threadIdx.x >> 5);
  const int tile_n = wid % (DMODEL / 64);
  const int tile_m = wid / (DMODEL / 64);
  const int m0 = tile_m * 32, n0 = tile_n * 64;
  const int lane = threadIdx.x & 31, hi = lane >> 4, nn = lane & 15;

  v8f acc[2][4];
#pragma unroll
  for (int mi = 0; mi < 2; ++mi)
#pragma unroll
    for (int f = 0; f < 4; ++f)
      acc[mi][f] = (v8f){0.f,0.f,0.f,0.f,0.f,0.f,0.f,0.f};

#pragma unroll 2
  for (int kk = 0; kk < INNER; kk += 32) {
    v16bf a0 = load_fragA(ctx, INNER, m0, kk);
    v16bf a1 = load_fragA(ctx, INNER, m0 + 16, kk);
    v16bf b0 = load_fragB_T(woutT, INNER, kk, n0);
    v16bf b1 = load_fragB_T(woutT, INNER, kk, n0 + 16);
    v16bf b2 = load_fragB_T(woutT, INNER, kk, n0 + 32);
    v16bf b3 = load_fragB_T(woutT, INNER, kk, n0 + 48);
    acc[0][0] = wmma_bf16(a0, b0, acc[0][0]);
    acc[0][1] = wmma_bf16(a0, b1, acc[0][1]);
    acc[0][2] = wmma_bf16(a0, b2, acc[0][2]);
    acc[0][3] = wmma_bf16(a0, b3, acc[0][3]);
    acc[1][0] = wmma_bf16(a1, b0, acc[1][0]);
    acc[1][1] = wmma_bf16(a1, b1, acc[1][1]);
    acc[1][2] = wmma_bf16(a1, b2, acc[1][2]);
    acc[1][3] = wmma_bf16(a1, b3, acc[1][3]);
  }

#pragma unroll
  for (int f = 0; f < 4; ++f) {
    const int col = n0 + f * 16 + nn;
    const float bias = b_out[col];
#pragma unroll
    for (int mi = 0; mi < 2; ++mi)
#pragma unroll
      for (int r = 0; r < 8; ++r) {
        int row = m0 + mi * 16 + r + hi * 8;
        out[(size_t)row * DMODEL + col] = acc[mi][f][r] + bias;
      }
  }
}

// ---------------------------------------------------------------------------
extern "C" void kernel_launch(void* const* d_in, const int* in_sizes, int n_in,
                              void* d_out, int out_size, void* d_ws, size_t ws_size,
                              hipStream_t stream) {
  const float* x     = (const float*)d_in[0];
  const float* W_qkv = (const float*)d_in[1];
  const float* W_out = (const float*)d_in[2];
  const float* b_out = (const float*)d_in[3];
  const float* temp  = (const float*)d_in[4];
  float* out = (float*)d_out;

  char* w = (char*)d_ws;
  size_t off = 0;
  auto alloc_bf16 = [&](size_t elems) -> __bf16* {
    __bf16* p = (__bf16*)(w + off);
    off = (off + elems * sizeof(__bf16) + 255) & ~(size_t)255;
    return p;
  };
  __bf16* xb    = alloc_bf16((size_t)ROWS * DMODEL);     // 16 MB
  __bf16* wqkvT = alloc_bf16((size_t)E3 * DMODEL);       // 6 MB  [3072][1024]
  __bf16* woutT = alloc_bf16((size_t)DMODEL * INNER);    // 2 MB  [1024][1024]
  __bf16* qb    = alloc_bf16((size_t)BATCH * HEADS * SEQ * DHEAD);   // 16 MB
  __bf16* kb    = alloc_bf16((size_t)BATCH * HEADS * SEQ * DHEAD);   // 16 MB
  __bf16* vT    = alloc_bf16((size_t)BATCH * HEADS * DHEAD * SEQ);   // 16 MB
  __bf16* ctx   = alloc_bf16((size_t)ROWS * INNER);                  // 16 MB
  (void)ws_size; (void)in_sizes; (void)n_in; (void)out_size;

  // 1) precision conversion / weight transposes
  cvt_bf16_kernel<<<4096, 256, 0, stream>>>(x, xb, ROWS * DMODEL);
  cvt_bf16_T_kernel<<<2048, 256, 0, stream>>>(W_qkv, wqkvT, DMODEL, E3);
  cvt_bf16_T_kernel<<<1024, 256, 0, stream>>>(W_out, woutT, INNER, DMODEL);

  // 2) QKV projection: (8192/32)*(3072/64) = 12288 wave-tiles, 8 waves/block
  qkv_gemm_kernel<<<12288 / 8, 256, 0, stream>>>(xb, wqkvT, temp, qb, kb, vT);

  // 3) attention: 64 (b,h) * 128 q-tiles = 8192 waves
  attn_kernel<<<8192 / 8, 256, 0, stream>>>(qb, kb, vT, ctx);

  // 4) output projection: (8192/32)*(1024/64) = 4096 wave-tiles
  out_gemm_kernel<<<4096 / 8, 256, 0, stream>>>(ctx, woutT, b_out, out);
}
